// EquivariantInteractionBlock_71983651881128
// MI455X (gfx1250) — matrix operand order repaired and checked by reading, not compiled, not run on hardware
//
#include <hip/hip_runtime.h>
#include <math.h>

typedef _Float16 v16h __attribute__((ext_vector_type(16)));
typedef _Float16 v8h  __attribute__((ext_vector_type(8)));
typedef float    v8f  __attribute__((ext_vector_type(8)));
typedef float    v4f  __attribute__((ext_vector_type(4)));

#define WPB 4                       // waves per block (wave32)
#define NRBF 32
#define INV24F 0.20412414523193154f // 1/sqrt(24)
#define INV72F 0.11785113019775793f // 1/sqrt(72)

// fast sigmoid: v_exp_f32 + v_rcp_f32 (no IEEE div chain)
__device__ __forceinline__ float sigmoidf_(float x) {
    return __builtin_amdgcn_rcpf(1.0f + __expf(-x));
}

// ---------------------------------------------------------------------------
// Kernel 0: zero output accumulator + BN-stat scratch
// ---------------------------------------------------------------------------
__global__ void zero_kernel(float* __restrict__ out, int n, float* __restrict__ stats) {
    for (int i = blockIdx.x * blockDim.x + threadIdx.x; i < n; i += gridDim.x * blockDim.x)
        out[i] = 0.0f;
    if (blockIdx.x == 0 && threadIdx.x < 48) stats[threadIdx.x] = 0.0f;
}

// ---------------------------------------------------------------------------
// Kernel 1: fused edge pipeline: RBF -> MLP (WMMA f16) -> equivariant message
//           -> atomic scatter into out (acts as agg buffer)
// One wave32 processes a tile of 16 edges.
// ---------------------------------------------------------------------------
__global__ __launch_bounds__(WPB * 32)
void edge_kernel(const float* __restrict__ x,
                 const int*   __restrict__ eidx,   // [2][E]
                 const float* __restrict__ evec,   // [E][3]
                 const float* __restrict__ edist,  // [E]
                 const float* __restrict__ W1, const float* __restrict__ b1,   // 32x64
                 const float* __restrict__ W2, const float* __restrict__ b2,   // 64x40
                 const float* __restrict__ W3, const float* __restrict__ b3,   // 40x576
                 float* __restrict__ out, int E)
{
    // ---- pre-packed WMMA B fragments (f16): [frag][lane][16 halves] ----
    __shared__ __align__(32) _Float16 pW1[4][32][16];    // K=32, N=64  (4 n-tiles)
    __shared__ __align__(32) _Float16 pW2[6][32][16];    // K=64, Npad=48 (2 ks x 3 nt)
    __shared__ __align__(32) _Float16 pW3[72][32][16];   // Kpad=64, N=576 (2 ks x 36 nt)
    __shared__ float sb1[64];
    __shared__ float sb2[48];
    __shared__ float sb3[576];
    // ---- per-wave edge-tile staging ----
    __shared__ float dArr[WPB][16];
    __shared__ float validArr[WPB][16];
    __shared__ int   dstArr[WPB][16];
    __shared__ float y1Arr[WPB][16][3];
    __shared__ float sArr[WPB][16][16];
    __shared__ float vArr[WPB][16][24];
    __shared__ float dotArr[WPB][16][8];
    __shared__ __align__(16) _Float16 hA [WPB][16][64];  // layer-1 output (A for layer 2)
    __shared__ __align__(16) _Float16 h2A[WPB][16][64];  // layer-2 output (A for layer 3, cols 48..63 = 0)
    __shared__ __align__(16) float    wT [WPB][36][32][8]; // layer-3 C fragments (w tile)

    const int tid = threadIdx.x;
    const int w   = tid >> 5;
    const int l   = tid & 31;
    const int m   = l & 15;   // row / edge-in-tile
    const int hi  = l >> 4;   // lane half

    // ------------------ one-time per-block weight prepack -------------------
    for (int idx = tid; idx < (4 + 6 + 72) * 512; idx += WPB * 32) {
        const int f   = idx >> 9;
        const int rem = idx & 511;
        const int ln  = rem >> 4;
        const int j   = rem & 15;
        const int lhi = ln >> 4;
        const int c16 = ln & 15;
        if (f < 4) {                       // W1: B frag K = j + 16*lhi
            const int k = j + 16 * lhi;
            const int c = f * 16 + c16;
            pW1[f][ln][j] = (_Float16)W1[k * 64 + c];
        } else if (f < 10) {               // W2
            const int ff = f - 4, ks = ff / 3, nt = ff % 3;
            const int k = ks * 32 + j + 16 * lhi;
            const int c = nt * 16 + c16;
            pW2[ff][ln][j] = (_Float16)((c < 40) ? W2[k * 40 + c] : 0.0f);
        } else {                           // W3 (K padded 40->64 with zeros)
            const int ff = f - 10, ks = ff / 36, nt = ff % 36;
            const int k = ks * 32 + j + 16 * lhi;
            const int c = nt * 16 + c16;
            pW3[ff][ln][j] = (_Float16)((k < 40) ? W3[k * 576 + c] : 0.0f);
        }
    }
    for (int i = tid; i < 64;  i += WPB * 32) sb1[i] = b1[i];
    for (int i = tid; i < 48;  i += WPB * 32) sb2[i] = (i < 40) ? b2[i] : 0.0f;
    for (int i = tid; i < 576; i += WPB * 32) sb3[i] = b3[i];
    // zero the K-padding columns of h2A once (never written again)
    for (int i = l; i < 256; i += 32) h2A[w][i >> 4][48 + (i & 15)] = (_Float16)0.0f;
    __syncthreads();

    const float gamma = 38.44f;               // ((NRBF-1)/CUTOFF)^2
    const float cstep = 5.0f / (NRBF - 1);
    const int nTiles = (E + 15) >> 4;
    const int waveG = blockIdx.x * WPB + w;
    const int waveStride = gridDim.x * WPB;

    for (int tile = waveG; tile < nTiles; tile += waveStride) {
        const int base = tile << 4;
        const int e    = base + m;
        const bool inb = (e < E);

        // ---------------- edge-data gather (lanes split work) ----------------
        int sidx = 0;
        if (hi == 0) {
            const float d  = inb ? edist[e] : 0.0f;
            sidx           = inb ? eidx[e] : 0;
            const int didx = inb ? eidx[E + e] : 0;
            float vx = 0.f, vy = 0.f, vz = 0.f;
            if (inb) { vx = evec[3 * e]; vy = evec[3 * e + 1]; vz = evec[3 * e + 2]; }
            const float nrm = sqrtf(vx * vx + vy * vy + vz * vz);
            const float inv = 1.7320508075688772f * __builtin_amdgcn_rcpf(fmaxf(nrm, 1e-12f));
            dArr[w][m] = d;
            dstArr[w][m] = didx;
            validArr[w][m] = (inb && d > 1e-8f) ? 1.0f : 0.0f;
            y1Arr[w][m][0] = vx * inv; y1Arr[w][m][1] = vy * inv; y1Arr[w][m][2] = vz * inv;
            const float* xp = x + (long)sidx * 40;
            #pragma unroll
            for (int i = 0; i < 16; i++) sArr[w][m][i] = inb ? xp[i] : 0.0f;
        }
        const int sidx_b = __shfl(sidx, m, 32);
        if (hi == 1) {
            const float* xp = x + (long)sidx_b * 40 + 16;
            float vv[24];
            #pragma unroll
            for (int i = 0; i < 24; i++) { vv[i] = inb ? xp[i] : 0.0f; vArr[w][m][i] = vv[i]; }
            const float y0 = y1Arr[w][m][0], y1v = y1Arr[w][m][1], y2 = y1Arr[w][m][2];
            #pragma unroll
            for (int u = 0; u < 8; u++)
                dotArr[w][m][u] = vv[3 * u] * y0 + vv[3 * u + 1] * y1v + vv[3 * u + 2] * y2;
        }

        // ------------- layer 1: build phi A-fragment in registers -------------
        const float dloc = dArr[w][m];
        v16h a1;
        #pragma unroll
        for (int j = 0; j < 16; j++) {
            const int k = j + 16 * hi;   // A 16x32 f16 layout (contiguous k = 16h + s)
            const float t = dloc - cstep * (float)k;
            a1[j] = (_Float16)__expf(-gamma * t * t);
        }
        #pragma unroll
        for (int nt = 0; nt < 4; nt++) {
            const v16h bf = *(const v16h*)&pW1[nt][l][0];
            v8f c = {};
            c = __builtin_amdgcn_wmma_f32_16x16x32_f16(false, a1, false, bf, (short)0, c, false, false);
            const int col = nt * 16 + m;
            const float bb = sb1[col];
            #pragma unroll
            for (int r = 0; r < 8; r++) {
                const float xv_ = c[r] + bb;
                hA[w][r + 8 * hi][col] = (_Float16)(xv_ * sigmoidf_(xv_));   // silu
            }
        }

        // ------------------------ layer 2 (K=64, N=48) ------------------------
        v16h a2[2];
        #pragma unroll
        for (int ks = 0; ks < 2; ks++) {
            const _Float16* hrow = &hA[w][m][0];
            const v8h lo = *(const v8h*)(hrow + ks * 32 + 16 * hi);
            const v8h hh = *(const v8h*)(hrow + ks * 32 + 8 + 16 * hi);
            #pragma unroll
            for (int i = 0; i < 8; i++) { a2[ks][i] = lo[i]; a2[ks][i + 8] = hh[i]; }
        }
        #pragma unroll
        for (int nt = 0; nt < 3; nt++) {
            v8f c = {};
            #pragma unroll
            for (int ks = 0; ks < 2; ks++) {
                const v16h bf = *(const v16h*)&pW2[ks * 3 + nt][l][0];
                c = __builtin_amdgcn_wmma_f32_16x16x32_f16(false, a2[ks], false, bf, (short)0, c, false, false);
            }
            const int col = nt * 16 + m;
            const float bb = sb2[col];
            #pragma unroll
            for (int r = 0; r < 8; r++) {
                const float xv_ = c[r] + bb;
                h2A[w][r + 8 * hi][col] = (_Float16)(xv_ * sigmoidf_(xv_));
            }
        }

        // ---------------------- layer 3 (K=64, N=576) -------------------------
        v16h a3[2];
        #pragma unroll
        for (int ks = 0; ks < 2; ks++) {
            const _Float16* hrow = &h2A[w][m][0];
            const v8h lo = *(const v8h*)(hrow + ks * 32 + 16 * hi);
            const v8h hh = *(const v8h*)(hrow + ks * 32 + 8 + 16 * hi);
            #pragma unroll
            for (int i = 0; i < 8; i++) { a3[ks][i] = lo[i]; a3[ks][i + 8] = hh[i]; }
        }
        #pragma unroll
        for (int nt = 0; nt < 36; nt++) {
            v8f c = {};
            #pragma unroll
            for (int ks = 0; ks < 2; ks++) {
                const v16h bf = *(const v16h*)&pW3[ks * 36 + nt][l][0];
                c = __builtin_amdgcn_wmma_f32_16x16x32_f16(false, a3[ks], false, bf, (short)0, c, false, false);
            }
            const float bb = sb3[nt * 16 + m];
            v4f lo4, hi4;
            #pragma unroll
            for (int r = 0; r < 4; r++) { lo4[r] = c[r] + bb; hi4[r] = c[r + 4] + bb; }
            *(v4f*)&wT[w][nt][l][0] = lo4;   // fragment-major: w[e=r+8h][16nt+n] = wT[nt][n+16h][r]
            *(v4f*)&wT[w][nt][l][4] = hi4;
        }

        // --------------- equivariant message + atomic scatter -----------------
        {
            const int eL = m;                        // 2 lanes per edge (hi = part)
            const int l2b = (eL >= 8) ? 16 : 0;
            const int r   = eL & 7;
            const float scale = validArr[w][eL];
            const int dn = dstArr[w][eL];
            float* op = out + (long)dn * 40;
            float su[16];
            #pragma unroll
            for (int u = 0; u < 16; u++) su[u] = sArr[w][eL][u];

            if (hi == 0) {
                // msg_s[wo] = INV24 * sum_u Wss[u,wo] s[u] + INV72 * sum_u Wvs[u,wo] dot[u]
                float du[8];
                #pragma unroll
                for (int u = 0; u < 8; u++) du[u] = dotArr[w][eL][u];
                #pragma unroll
                for (int wo = 0; wo < 16; wo++) {
                    float acc = 0.0f, acc2 = 0.0f;
                    #pragma unroll
                    for (int u = 0; u < 16; u++)            // cols u*16+wo -> tile u, pos wo
                        acc += wT[w][u][wo + l2b][r] * su[u];
                    #pragma unroll
                    for (int u = 0; u < 8; u++)             // cols 256+u*16+wo -> tile 16+u
                        acc2 += wT[w][16 + u][wo + l2b][r] * du[u];
                    const float msg = INV24F * acc + INV72F * acc2;
                    if (scale != 0.0f) atomicAdd(op + wo, msg);
                }
            } else {
                // msg_v[wo][d] = INV24 * ( (sum_u Wsv[u,wo] s[u]) * y1[d] + sum_u Wvv[u,wo] v[u][d] )
                const float y0 = y1Arr[w][eL][0], y1v = y1Arr[w][eL][1], y2 = y1Arr[w][eL][2];
                float vv[24];
                #pragma unroll
                for (int i = 0; i < 24; i++) vv[i] = vArr[w][eL][i];
                #pragma unroll
                for (int wo = 0; wo < 8; wo++) {
                    float ts = 0.0f;
                    #pragma unroll
                    for (int u = 0; u < 16; u++) {
                        const int cc = 384 + u * 8 + wo;    // W_sv region
                        ts += wT[w][cc >> 4][(cc & 15) + l2b][r] * su[u];
                    }
                    float a0 = 0.f, a1v = 0.f, a2v = 0.f;
                    #pragma unroll
                    for (int u = 0; u < 8; u++) {
                        const int cc = 512 + u * 8 + wo;    // W_vv region
                        const float wv_ = wT[w][cc >> 4][(cc & 15) + l2b][r];
                        a0 += wv_ * vv[3 * u]; a1v += wv_ * vv[3 * u + 1]; a2v += wv_ * vv[3 * u + 2];
                    }
                    const float m0 = INV24F * (ts * y0  + a0);
                    const float m1 = INV24F * (ts * y1v + a1v);
                    const float m2 = INV24F * (ts * y2  + a2v);
                    if (scale != 0.0f) {
                        atomicAdd(op + 16 + wo * 3 + 0, m0);
                        atomicAdd(op + 16 + wo * 3 + 1, m1);
                        atomicAdd(op + 16 + wo * 3 + 2, m2);
                    }
                }
            }
        }
    }
}

// ---------------------------------------------------------------------------
// Kernel 2: u = self-transform + agg (in-place in out) + BN stat accumulation
// ---------------------------------------------------------------------------
__global__ __launch_bounds__(256)
void node_pass1(const float* __restrict__ x,
                const float* __restrict__ Ws, const float* __restrict__ Wv,
                float* __restrict__ out, float* __restrict__ stats, int N)
{
    __shared__ float sWs[256];
    __shared__ float sWv[64];
    const int tid = threadIdx.x;
    if (tid < 256) sWs[tid] = Ws[tid];
    if (tid < 64)  sWv[tid] = Wv[tid];
    __syncthreads();

    const int n = blockIdx.x * 256 + tid;
    float us[16], uv[24];
    if (n < N) {
        const float* xp = x + (long)n * 40;
        float* op = out + (long)n * 40;
        float xs[16], xv[24];
        #pragma unroll
        for (int i = 0; i < 16; i++) xs[i] = xp[i];
        #pragma unroll
        for (int i = 0; i < 24; i++) xv[i] = xp[16 + i];
        #pragma unroll
        for (int i = 0; i < 16; i++) {
            float acc = 0.0f;
            #pragma unroll
            for (int j = 0; j < 16; j++) acc += sWs[i * 16 + j] * xs[j];   // xs @ Ws^T
            us[i] = acc * 0.25f + op[i];
        }
        #pragma unroll
        for (int vo = 0; vo < 8; vo++) {
            float a0 = 0.f, a1 = 0.f, a2 = 0.f;
            #pragma unroll
            for (int u = 0; u < 8; u++) {
                const float wv_ = sWv[vo * 8 + u];
                a0 += wv_ * xv[3 * u]; a1 += wv_ * xv[3 * u + 1]; a2 += wv_ * xv[3 * u + 2];
            }
            uv[vo * 3 + 0] = a0 * 0.3535533905932738f + op[16 + vo * 3 + 0];
            uv[vo * 3 + 1] = a1 * 0.3535533905932738f + op[16 + vo * 3 + 1];
            uv[vo * 3 + 2] = a2 * 0.3535533905932738f + op[16 + vo * 3 + 2];
        }
        #pragma unroll
        for (int i = 0; i < 16; i++) op[i] = us[i];
        #pragma unroll
        for (int i = 0; i < 24; i++) op[16 + i] = uv[i];
    } else {
        #pragma unroll
        for (int i = 0; i < 16; i++) us[i] = 0.0f;
        #pragma unroll
        for (int i = 0; i < 24; i++) uv[i] = 0.0f;
    }

    // stats: [0..15] sum_s, [16..31] sumsq_s, [32..39] sum over d of u_v^2
    float part[40];
    #pragma unroll
    for (int i = 0; i < 16; i++) { part[i] = us[i]; part[16 + i] = us[i] * us[i]; }
    #pragma unroll
    for (int w8 = 0; w8 < 8; w8++)
        part[32 + w8] = uv[3 * w8] * uv[3 * w8] + uv[3 * w8 + 1] * uv[3 * w8 + 1]
                      + uv[3 * w8 + 2] * uv[3 * w8 + 2];
    #pragma unroll
    for (int i = 0; i < 40; i++) {
        float v = part[i];
        for (int off = 16; off > 0; off >>= 1) v += __shfl_xor(v, off, 32);
        if ((tid & 31) == 0) atomicAdd(&stats[i], v);
    }
}

// ---------------------------------------------------------------------------
// Kernel 3: batch-norm + norm-gate nonlinearity (silu(n)/n == sigmoid(n))
// ---------------------------------------------------------------------------
__global__ __launch_bounds__(256)
void node_pass2(const float* __restrict__ stats,
                const float* __restrict__ bnw, const float* __restrict__ bnb,
                float* __restrict__ out, int N)
{
    const int n = blockIdx.x * 256 + threadIdx.x;
    if (n >= N) return;
    const float invN = 1.0f / (float)N;
    float* op = out + (long)n * 40;
    #pragma unroll
    for (int i = 0; i < 16; i++) {
        const float mean = stats[i] * invN;
        const float var  = fmaxf(stats[16 + i] * invN - mean * mean, 0.0f);
        const float sc   = __builtin_amdgcn_rsqf(var + 1e-5f) * bnw[i];
        const float cs   = (op[i] - mean) * sc + bnb[i];
        const float ns   = sqrtf(cs * cs + 1e-12f);
        op[i] = cs * sigmoidf_(ns);
    }
    #pragma unroll
    for (int w8 = 0; w8 < 8; w8++) {
        const float vn = stats[32 + w8] * (invN / 3.0f);
        const float sc = __builtin_amdgcn_rsqf(vn + 1e-5f) * bnw[16 + w8];
        const float c0 = op[16 + w8 * 3 + 0] * sc;
        const float c1 = op[16 + w8 * 3 + 1] * sc;
        const float c2 = op[16 + w8 * 3 + 2] * sc;
        const float nv = sqrtf(c0 * c0 + c1 * c1 + c2 * c2 + 1e-12f);
        const float g  = sigmoidf_(nv);
        op[16 + w8 * 3 + 0] = c0 * g;
        op[16 + w8 * 3 + 1] = c1 * g;
        op[16 + w8 * 3 + 2] = c2 * g;
    }
}

// ---------------------------------------------------------------------------
extern "C" void kernel_launch(void* const* d_in, const int* in_sizes, int n_in,
                              void* d_out, int out_size, void* d_ws, size_t ws_size,
                              hipStream_t stream) {
    const float* x     = (const float*)d_in[0];
    const int*   eidx  = (const int*)  d_in[1];
    const float* evec  = (const float*)d_in[2];
    const float* edist = (const float*)d_in[3];
    const float* W1    = (const float*)d_in[4];
    const float* b1    = (const float*)d_in[5];
    const float* W2    = (const float*)d_in[6];
    const float* b2    = (const float*)d_in[7];
    const float* W3    = (const float*)d_in[8];
    const float* b3    = (const float*)d_in[9];
    const float* Ws    = (const float*)d_in[10];
    const float* Wv    = (const float*)d_in[11];
    const float* bnw   = (const float*)d_in[12];
    const float* bnb   = (const float*)d_in[13];
    float* out   = (float*)d_out;
    float* stats = (float*)d_ws;

    const int E = in_sizes[3];        // edge_distance length
    const int N = in_sizes[0] / 40;   // x is (N, 40)

    zero_kernel<<<512, 256, 0, stream>>>(out, out_size, stats);
    edge_kernel<<<1024, WPB * 32, 0, stream>>>(x, eidx, evec, edist,
                                               W1, b1, W2, b2, W3, b3, out, E);
    const int nb = (N + 255) / 256;
    node_pass1<<<nb, 256, 0, stream>>>(x, Ws, Wv, out, stats, N);
    node_pass2<<<nb, 256, 0, stream>>>(stats, bnw, bnb, out, N);
}
